// GNNEncoder_42374147342606
// MI455X (gfx1250) — matrix operand order repaired and checked by reading, not compiled, not run on hardware
//
#include <hip/hip_runtime.h>

typedef __attribute__((ext_vector_type(16))) __bf16       v16bf;
typedef __attribute__((ext_vector_type(8)))  float        v8f;
typedef __attribute__((ext_vector_type(4)))  unsigned int u32x4;

#define F_DIM 256
#define EPB   8

__device__ __forceinline__ unsigned short f2bf(float x) {
    unsigned int u = __float_as_uint(x);
    u += 0x7FFFu + ((u >> 16) & 1u);   // round-to-nearest-even
    return (unsigned short)(u >> 16);
}

__device__ __forceinline__ float wave_sum(float v) {
#pragma unroll
    for (int m = 16; m; m >>= 1) v += __shfl_xor(v, m, 32);
    return v;
}

__global__ void cvt_f32_bf16(const float* __restrict__ in, unsigned short* __restrict__ out, int n) {
    int i = blockIdx.x * blockDim.x + threadIdx.x;
    if (i < n) out[i] = f2bf(in[i]);
}

// Wt[F][K] = bf16(W[K][F])^T : makes each lane's WMMA B-fragment contiguous (2x b128 loads)
__global__ void transpose_bf16(const float* __restrict__ W, unsigned short* __restrict__ Wt,
                               int K, int F) {
    int idx = blockIdx.x * blockDim.x + threadIdx.x;
    if (idx < K * F) {
        int k = idx % K, n = idx / K;
        Wt[(size_t)n * K + k] = f2bf(W[(size_t)k * F + n]);
    }
}

__global__ void degree_count(const int* __restrict__ dst, float* __restrict__ deg, int nE) {
    int e = blockIdx.x * blockDim.x + threadIdx.x;
    if (e < nE) atomicAdd(&deg[dst[e]], 1.0f);
}

__global__ void inv_sqrt_deg(const float* __restrict__ deg, float* __restrict__ dis, int n) {
    int i = blockIdx.x * blockDim.x + threadIdx.x;
    if (i < n) dis[i] = rsqrtf(deg[i] + 1.0f);
}

// H[N,F] = bf16(X)[N,K] @ bf16(W)[K,F] via WMMA; 4 row-tiles per wave share one B fragment.
// Fast path (all 4 tiles valid) is branch-free with immediate-offset addressing.
template <int K>
__global__ void gemm_wmma_bf16(const unsigned short* __restrict__ Xb,
                               const unsigned short* __restrict__ Wt,  // [F][K] bf16
                               float* __restrict__ H, int N) {
    constexpr int F  = F_DIM;
    constexpr int TM = 4;
    const int wave = threadIdx.x >> 5;
    const int lane = threadIdx.x & 31;
    const int half = lane >> 4;            // 0: lanes 0-15, 1: lanes 16-31
    const int l16  = lane & 15;
    const int nTiles = (N + 15) >> 4;
    const int tile0  = (blockIdx.y * 8 + wave) * TM;
    if (tile0 >= nTiles) return;           // wave-uniform
    const int tileN = blockIdx.x;

    union U { v16bf v; u32x4 q[2]; };
    // Base pointers pre-biased by the lane-half offsets; all loop offsets are constants.
    const unsigned short* brow = Wt + (size_t)(tileN * 16 + l16) * K + 16 * half;
    const unsigned short* arow = Xb + (size_t)(tile0 * 16 + l16) * K + 16 * half;
    float* outb = H + (size_t)(tile0 * 16 + 8 * half) * F + tileN * 16 + l16;

    if (tile0 + TM <= nTiles) {            // wave-uniform fast path
        v8f c[TM] = {};
#pragma unroll
        for (int k0 = 0; k0 < K; k0 += 32) {
            U B;                            // elem j -> K = k0 + 16*half + j
            B.q[0] = *(const u32x4*)(brow + k0);
            B.q[1] = *(const u32x4*)(brow + k0 + 8);
#pragma unroll
            for (int t = 0; t < TM; ++t) {
                U A;                        // elem j -> K = k0 + 16*half + j
                A.q[0] = *(const u32x4*)(arow + t * 16 * K + k0);
                A.q[1] = *(const u32x4*)(arow + t * 16 * K + k0 + 8);
                c[t] = __builtin_amdgcn_wmma_f32_16x16x32_bf16(false, A.v, false, B.v,
                                                               (short)0, c[t], false, false);
            }
        }
#pragma unroll
        for (int t = 0; t < TM; ++t)
#pragma unroll
            for (int r = 0; r < 8; ++r)
                outb[(size_t)(t * 16 + r) * F] = c[t][r];
    } else {
        const int rem = nTiles - tile0;    // 1..TM-1, wave-uniform
        for (int t = 0; t < rem; ++t) {
            v8f c = {};
#pragma unroll
            for (int k0 = 0; k0 < K; k0 += 32) {
                U A, B;
                B.q[0] = *(const u32x4*)(brow + k0);
                B.q[1] = *(const u32x4*)(brow + k0 + 8);
                A.q[0] = *(const u32x4*)(arow + (size_t)t * 16 * K + k0);
                A.q[1] = *(const u32x4*)(arow + (size_t)t * 16 * K + k0 + 8);
                c = __builtin_amdgcn_wmma_f32_16x16x32_bf16(false, A.v, false, B.v,
                                                            (short)0, c, false, false);
            }
#pragma unroll
            for (int r = 0; r < 8; ++r)
                outb[(size_t)(t * 16 + r) * F] = c[r];
        }
    }
}

// agg[dst] += H[src] * dis[src]*dis[dst]; edge metadata staged in LDS once per block.
__global__ void edge_aggregate(const float* __restrict__ H, const int* __restrict__ src,
                               const int* __restrict__ dst, const float* __restrict__ dis,
                               float* __restrict__ agg, int nE) {
    __shared__ int   ss[EPB];
    __shared__ int   sd[EPB];
    __shared__ float sw[EPB];
    const int t  = threadIdx.x;
    const int e0 = blockIdx.x * EPB;
    if (t < EPB) {
        int e = e0 + t;
        if (e < nE) {
            int s = src[e], d = dst[e];
            ss[t] = s; sd[t] = d; sw[t] = dis[s] * dis[d];
        } else {
            sd[t] = -1;
        }
    }
    __syncthreads();
#pragma unroll
    for (int i = 0; i < EPB; ++i) {
        int d = sd[i];
        if (d < 0) continue;               // block-uniform
        atomicAdd(&agg[(size_t)d * F_DIM + t], H[(size_t)ss[i] * F_DIM + t] * sw[i]);
    }
}

// out = LN(relu(agg + H*dis^2 + bias)) * g + be, emitted as bf16 (next layer's GEMM A operand)
__global__ void fuse_relu_ln_bf16(const float* __restrict__ agg, const float* __restrict__ H,
                                  const float* __restrict__ bias, const float* __restrict__ g,
                                  const float* __restrict__ be, const float* __restrict__ dis,
                                  unsigned short* __restrict__ out) {
    const int i = blockIdx.x, f = threadIdx.x;
    const int w = f >> 5, l = f & 31;
    __shared__ float red[F_DIM / 32];
    float dsl = dis[i];
    float v = agg[(size_t)i * F_DIM + f] + H[(size_t)i * F_DIM + f] * dsl * dsl + bias[f];
    v = fmaxf(v, 0.0f);

    float s = wave_sum(v);
    if (l == 0) red[w] = s;
    __syncthreads();
    float mu = 0.0f;
#pragma unroll
    for (int j = 0; j < F_DIM / 32; ++j) mu += red[j];
    mu *= (1.0f / F_DIM);
    __syncthreads();

    float dv = v - mu;
    float s2 = wave_sum(dv * dv);
    if (l == 0) red[w] = s2;
    __syncthreads();
    float var = 0.0f;
#pragma unroll
    for (int j = 0; j < F_DIM / 32; ++j) var += red[j];
    var *= (1.0f / F_DIM);

    float o = dv * rsqrtf(var + 1e-5f) * g[f] + be[f];
    out[(size_t)i * F_DIM + f] = f2bf(o);
}

// layer 3: agg += H*dis^2 + bias (fp32, in place)
__global__ void fuse_bias_self(const float* __restrict__ H, const float* __restrict__ bias,
                               const float* __restrict__ dis, float* __restrict__ agg) {
    const int i = blockIdx.x, f = threadIdx.x;
    float dsl = dis[i];
    agg[(size_t)i * F_DIM + f] += H[(size_t)i * F_DIM + f] * dsl * dsl + bias[f];
}

__global__ void pool_accum(const float* __restrict__ h, const int* __restrict__ batch,
                           float* __restrict__ pool, float* __restrict__ cnt) {
    const int i = blockIdx.x, f = threadIdx.x;
    int b = batch[i];
    atomicAdd(&pool[(size_t)b * F_DIM + f], h[(size_t)i * F_DIM + f]);
    if (f == 0) atomicAdd(&cnt[b], 1.0f);
}

__global__ void pool_finalize(const float* __restrict__ pool, const float* __restrict__ cnt,
                              float* __restrict__ out) {
    const int g = blockIdx.x, f = threadIdx.x;
    out[(size_t)g * F_DIM + f] = pool[(size_t)g * F_DIM + f] / cnt[g];
}

extern "C" void kernel_launch(void* const* d_in, const int* in_sizes, int n_in,
                              void* d_out, int out_size, void* d_ws, size_t ws_size,
                              hipStream_t stream) {
    const float* x   = (const float*)d_in[0];
    const float* W1  = (const float*)d_in[1];
    const float* b1  = (const float*)d_in[2];
    const float* g1  = (const float*)d_in[3];
    const float* be1 = (const float*)d_in[4];
    const float* W2  = (const float*)d_in[5];
    const float* b2  = (const float*)d_in[6];
    const float* g2  = (const float*)d_in[7];
    const float* be2 = (const float*)d_in[8];
    const float* W3  = (const float*)d_in[9];
    const float* b3  = (const float*)d_in[10];
    const int* eidx  = (const int*)d_in[11];
    const int* batch = (const int*)d_in[12];

    const int IN = 128, HID = 256, G = 256;
    const int N  = in_sizes[0] / IN;
    const int nE = in_sizes[11] / 2;
    const int* src = eidx;
    const int* dst = eidx + nE;

    char* ws = (char*)d_ws;
    size_t off = 0;
    auto alloc = [&](size_t bytes) -> char* {
        char* p = ws + off;
        off = (off + bytes + 255) & ~(size_t)255;
        return p;
    };
    unsigned short* Wt  = (unsigned short*)alloc((size_t)HID * HID * 2);  // transposed bf16 W
    unsigned short* Xb  = (unsigned short*)alloc((size_t)N * HID * 2);
    float* H    = (float*)alloc((size_t)N * HID * 4);
    float* AGG  = (float*)alloc((size_t)N * HID * 4);
    float* deg  = (float*)alloc((size_t)N * 4);
    float* dis  = (float*)alloc((size_t)N * 4);
    float* pool = (float*)alloc((size_t)G * HID * 4);
    float* cnt  = (float*)alloc((size_t)G * 4);
    (void)ws_size; (void)n_in; (void)out_size;

    dim3 blk(256);
    const int nTiles  = (N + 15) / 16;
    const int eBlocks = (nE + EPB - 1) / EPB;
    dim3 gemmGrid(HID / 16, (nTiles + 8 * 4 - 1) / (8 * 4));

    // ---- symmetric normalization (A + I) ----
    hipMemsetAsync(deg, 0, (size_t)N * 4, stream);
    degree_count<<<(nE + 255) / 256, blk, 0, stream>>>(dst, deg, nE);
    inv_sqrt_deg<<<(N + 255) / 256, blk, 0, stream>>>(deg, dis, N);

    // ---- layer 1 (K = 128) ----
    cvt_f32_bf16<<<(N * IN + 255) / 256, blk, 0, stream>>>(x, Xb, N * IN);
    transpose_bf16<<<(IN * HID + 255) / 256, blk, 0, stream>>>(W1, Wt, IN, HID);
    gemm_wmma_bf16<128><<<gemmGrid, blk, 0, stream>>>(Xb, Wt, H, N);
    hipMemsetAsync(AGG, 0, (size_t)N * HID * 4, stream);
    edge_aggregate<<<eBlocks, blk, 0, stream>>>(H, src, dst, dis, AGG, nE);
    fuse_relu_ln_bf16<<<N, blk, 0, stream>>>(AGG, H, b1, g1, be1, dis, Xb);

    // ---- layer 2 (K = 256) ----
    transpose_bf16<<<(HID * HID + 255) / 256, blk, 0, stream>>>(W2, Wt, HID, HID);
    gemm_wmma_bf16<256><<<gemmGrid, blk, 0, stream>>>(Xb, Wt, H, N);
    hipMemsetAsync(AGG, 0, (size_t)N * HID * 4, stream);
    edge_aggregate<<<eBlocks, blk, 0, stream>>>(H, src, dst, dis, AGG, nE);
    fuse_relu_ln_bf16<<<N, blk, 0, stream>>>(AGG, H, b2, g2, be2, dis, Xb);

    // ---- layer 3 (K = 256) ----
    transpose_bf16<<<(HID * HID + 255) / 256, blk, 0, stream>>>(W3, Wt, HID, HID);
    gemm_wmma_bf16<256><<<gemmGrid, blk, 0, stream>>>(Xb, Wt, H, N);
    hipMemsetAsync(AGG, 0, (size_t)N * HID * 4, stream);
    edge_aggregate<<<eBlocks, blk, 0, stream>>>(H, src, dst, dis, AGG, nE);
    fuse_bias_self<<<N, blk, 0, stream>>>(H, b3, dis, AGG);

    // ---- segment mean pool ----
    hipMemsetAsync(pool, 0, (size_t)G * HID * 4, stream);
    hipMemsetAsync(cnt, 0, (size_t)G * 4, stream);
    pool_accum<<<N, blk, 0, stream>>>(AGG, batch, pool, cnt);
    pool_finalize<<<G, blk, 0, stream>>>(pool, cnt, (float*)d_out);
}
